// CaNetConv_28948079575210
// MI455X (gfx1250) — compile-verified, matrix operand order checked by reading, not run on hardware
//
#include <hip/hip_runtime.h>
#include <math.h>

typedef _Float16 f16;
typedef _Float16 v16h __attribute__((ext_vector_type(16)));
typedef _Float16 v8h  __attribute__((ext_vector_type(8)));
typedef float    v8f  __attribute__((ext_vector_type(8)));

#define DFEAT 128
#define KH 4
#define NT 4   // 16-node tiles per block in the GEMM

// ---------------- init: out = x (residual), denom = 0, lmax = -inf ----------
__global__ void init_kernel(const float* __restrict__ x, float* __restrict__ out,
                            float* __restrict__ denom, float* __restrict__ lmax,
                            int Nn) {
    size_t i = (size_t)blockIdx.x * blockDim.x + threadIdx.x;
    size_t total = (size_t)Nn * DFEAT;
    if (i < total) out[i] = x[i];
    if (i < (size_t)KH * Nn) denom[i] = 0.0f;
    if (i < KH) lmax[i] = -__builtin_inff();
}

// ---------------- prep: x -> f16, W -> f16 transposed (wt[k][f][d]) ---------
__global__ void prep_kernel(const float* __restrict__ x, const float* __restrict__ w,
                            f16* __restrict__ xh, f16* __restrict__ wt, int Nn) {
    size_t i = (size_t)blockIdx.x * blockDim.x + threadIdx.x;
    if (i < (size_t)Nn * DFEAT) xh[i] = (f16)x[i];
    if (i < (size_t)KH * DFEAT * DFEAT) {
        int k = (int)(i >> 14);        // / (128*128)
        int d = (int)((i >> 7) & 127); // input dim
        int f = (int)(i & 127);        // output dim
        wt[((size_t)k * DFEAT + f) * DFEAT + d] = (f16)w[i];
    }
}

// ---------------- WMMA GEMM: h[k] = x @ W[k] ---------------------------------
// Operands swapped so D = Wt(16f x 32k) * x^T(32k x 16n):
//   M = feature, N = node  =>  lane (g,ml) holds node `ml`, features g*8+0..7
//   => accumulator packs to 8 consecutive f16 of h[k][node][*]: ONE b128 store.
// Block = 256 threads = 8 waves; wave w owns feature tile w (16 features).
// A (Wt) fragments loaded once per wave, reused across NT node tiles.
__global__ void __launch_bounds__(256) gemm_kernel(const f16* __restrict__ xh,
                                                   const f16* __restrict__ wt,
                                                   f16* __restrict__ h, int Nn) {
    const int nt0  = blockIdx.x * NT;   // first 16-node tile
    const int kk   = blockIdx.y;        // head
    const int wave = threadIdx.x >> 5;  // feature tile 0..7
    const int lane = threadIdx.x & 31;
    const int g    = lane >> 4;
    const int ml   = lane & 15;

    // A fragments: rows of Wt (feature f = wave*16 + ml), contiguous k-chunks
    const f16* wrow = wt + ((size_t)kk * DFEAT + (wave * 16 + ml)) * DFEAT;
    v16h A[4];
#pragma unroll
    for (int kt = 0; kt < 4; ++kt) {
        const int k0 = kt * 32 + g * 8;
        v8h lo = *(const v8h*)(wrow + k0);
        v8h hi = *(const v8h*)(wrow + k0 + 16);
        A[kt] = __builtin_shufflevector(lo, hi, 0,1,2,3,4,5,6,7,8,9,10,11,12,13,14,15);
    }

#pragma unroll
    for (int t = 0; t < NT; ++t) {
        const int node  = (nt0 + t) * 16 + ml;
        const int nodec = node < Nn ? node : Nn - 1;   // clamp: EXEC stays all-ones
        const f16* xrow = xh + (size_t)nodec * DFEAT;
        v8f c = {};
#pragma unroll
        for (int kt = 0; kt < 4; ++kt) {
            const int k0 = kt * 32 + g * 8;
            v8h lo = *(const v8h*)(xrow + k0);
            v8h hi = *(const v8h*)(xrow + k0 + 16);
            v16h B = __builtin_shufflevector(lo, hi, 0,1,2,3,4,5,6,7,8,9,10,11,12,13,14,15);
            c = __builtin_amdgcn_wmma_f32_16x16x32_f16(false, A[kt], false, B,
                                                       (short)0, c, false, false);
        }
        v8h pack;
#pragma unroll
        for (int r = 0; r < 8; ++r) pack[r] = (f16)c[r];
        if (node < Nn)
            *(v8h*)(h + ((size_t)kk * Nn + node) * DFEAT + wave * 16 + g * 8) = pack;
    }
}

// ---------------- scores: s_src/s_dst = h . a1 / a2 (wave per (k,n)) --------
__global__ void scores_kernel(const f16* __restrict__ h, const float* __restrict__ a,
                              float* __restrict__ s_src, float* __restrict__ s_dst,
                              int Nn) {
    const int lane = threadIdx.x & 31;
    const int wave = threadIdx.x >> 5;
    long idx = (long)blockIdx.x * 8 + wave;   // idx = k*Nn + n
    if (idx >= (long)KH * Nn) return;         // uniform per wave
    int k = (int)(idx / Nn);
    const f16*  hp = h + (size_t)idx * DFEAT;
    const float* a1 = a + (size_t)k * 2 * DFEAT;
    const float* a2 = a1 + DFEAT;
    float ssrc = 0.f, sdst = 0.f;
#pragma unroll
    for (int j = 0; j < 4; ++j) {
        int d = j * 32 + lane;
        float hv = (float)hp[d];
        ssrc += hv * a1[d];
        sdst += hv * a2[d];
    }
#pragma unroll
    for (int off = 16; off > 0; off >>= 1) {
        ssrc += __shfl_down(ssrc, off, 32);
        sdst += __shfl_down(sdst, off, 32);
    }
    if (lane == 0) { s_src[idx] = ssrc; s_dst[idx] = sdst; }
}

// ---------------- global per-head max of masked leaky-relu logits -----------
__device__ inline void atomicMaxFloat(float* addr, float val) {
    int* ia = (int*)addr;
    int old = *ia;
    while (__int_as_float(old) < val) {
        int assumed = old;
        old = atomicCAS(ia, assumed, __float_as_int(val));
        if (old == assumed) break;
    }
}

__global__ void max_kernel(const int* __restrict__ row, const int* __restrict__ col,
                           const float* __restrict__ s_src, const float* __restrict__ s_dst,
                           float* __restrict__ lmax, int Nn, int En) {
    __shared__ float sm[8][KH];
    int i = blockIdx.x * blockDim.x + threadIdx.x;
    float m[KH];
#pragma unroll
    for (int k = 0; k < KH; ++k) m[k] = -__builtin_inff();
    if (i < En + Nn) {
        int fr, fc; bool mask;
        if (i < En) { fr = row[i]; fc = col[i]; mask = (fr != fc); }
        else        { fr = fc = i - En; mask = true; }
        if (mask) {
#pragma unroll
            for (int k = 0; k < KH; ++k) {
                float v = s_src[(size_t)k * Nn + fr] + s_dst[(size_t)k * Nn + fc];
                m[k] = v > 0.f ? v : 0.01f * v;
            }
        }
    }
    const int lane = threadIdx.x & 31, wave = threadIdx.x >> 5;
#pragma unroll
    for (int off = 16; off > 0; off >>= 1)
#pragma unroll
        for (int k = 0; k < KH; ++k)
            m[k] = fmaxf(m[k], __shfl_down(m[k], off, 32));
    if (lane == 0)
#pragma unroll
        for (int k = 0; k < KH; ++k) sm[wave][k] = m[k];
    __syncthreads();
    if (threadIdx.x < KH) {
        float mm = -__builtin_inff();
#pragma unroll
        for (int wv = 0; wv < 8; ++wv) mm = fmaxf(mm, sm[wv][threadIdx.x]);
        atomicMaxFloat(&lmax[threadIdx.x], mm);
    }
}

// ---------------- denom: segment-sum of w over source nodes -----------------
__global__ void denom_kernel(const int* __restrict__ row, const int* __restrict__ col,
                             const float* __restrict__ s_src, const float* __restrict__ s_dst,
                             const float* __restrict__ lmax, float* __restrict__ denom,
                             int Nn, int En) {
    int i = blockIdx.x * blockDim.x + threadIdx.x;
    if (i >= En + Nn) return;
    int fr, fc; bool mask;
    if (i < En) { fr = row[i]; fc = col[i]; mask = (fr != fc); }
    else        { fr = fc = i - En; mask = true; }
    if (!mask) return;
#pragma unroll
    for (int k = 0; k < KH; ++k) {
        float v = s_src[(size_t)k * Nn + fr] + s_dst[(size_t)k * Nn + fc];
        v = v > 0.f ? v : 0.01f * v;
        atomicAdd(&denom[(size_t)k * Nn + fr], __expf(v - lmax[k]));
    }
}

// ---------------- scatter: out[fr] += sum_k coef_k * h[k][fc]  --------------
// One block (128 threads = feature dim) per edge; all 4 heads fused into a
// single fp32 atomic add per (edge, feature).
__global__ void scatter_kernel(const int* __restrict__ row, const int* __restrict__ col,
                               const f16* __restrict__ h,
                               const float* __restrict__ s_src, const float* __restrict__ s_dst,
                               const float* __restrict__ lmax, const float* __restrict__ denom,
                               const float* __restrict__ e, float* __restrict__ out,
                               int Nn, int En) {
    int ei = blockIdx.x;
    int fr, fc; bool mask;
    if (ei < En) { fr = row[ei]; fc = col[ei]; mask = (fr != fc); }
    else         { fr = fc = ei - En; mask = true; }
    if (!mask) return;
    int d = threadIdx.x;
    float acc = 0.f;
#pragma unroll
    for (int k = 0; k < KH; ++k) {
        float v = s_src[(size_t)k * Nn + fr] + s_dst[(size_t)k * Nn + fc];
        v = v > 0.f ? v : 0.01f * v;
        float wgt  = __expf(v - lmax[k]);
        float coef = wgt * e[(size_t)fr * KH + k] / (denom[(size_t)k * Nn + fr] + 1e-8f);
        acc += coef * (float)h[((size_t)k * Nn + fc) * DFEAT + d];
    }
    atomicAdd(&out[(size_t)fr * DFEAT + d], acc);
}

// ---------------------------------------------------------------------------
extern "C" void kernel_launch(void* const* d_in, const int* in_sizes, int n_in,
                              void* d_out, int out_size, void* d_ws, size_t ws_size,
                              hipStream_t stream) {
    const float* x   = (const float*)d_in[0];
    const int*   adj = (const int*)d_in[1];
    const float* e   = (const float*)d_in[2];
    const float* w   = (const float*)d_in[3];
    const float* a   = (const float*)d_in[4];
    float* out = (float*)d_out;

    const int Nn = in_sizes[0] / DFEAT;
    const int En = in_sizes[1] / 2;
    const int* rowp = adj;
    const int* colp = adj + En;

    // Workspace carve (≈ 67 MB total)
    char* ws = (char*)d_ws;
    size_t off = 0;
    auto carve = [&](size_t bytes) -> void* {
        void* p = ws + off;
        off += (bytes + 255) & ~(size_t)255;
        return p;
    };
    f16*   xh    = (f16*)  carve((size_t)Nn * DFEAT * sizeof(f16));
    f16*   wt    = (f16*)  carve((size_t)KH * DFEAT * DFEAT * sizeof(f16));
    f16*   h     = (f16*)  carve((size_t)KH * Nn * DFEAT * sizeof(f16));
    float* s_src = (float*)carve((size_t)KH * Nn * sizeof(float));
    float* s_dst = (float*)carve((size_t)KH * Nn * sizeof(float));
    float* denom = (float*)carve((size_t)KH * Nn * sizeof(float));
    float* lmax  = (float*)carve((size_t)KH * sizeof(float));

    size_t total = (size_t)Nn * DFEAT;
    int cblocks = (int)((total + 255) / 256);
    init_kernel<<<cblocks, 256, 0, stream>>>(x, out, denom, lmax, Nn);
    prep_kernel<<<cblocks, 256, 0, stream>>>(x, w, xh, wt, Nn);

    dim3 ggrid((Nn + NT * 16 - 1) / (NT * 16), KH);
    gemm_kernel<<<ggrid, 256, 0, stream>>>(xh, wt, h, Nn);

    int sblocks = (int)(((size_t)KH * Nn + 7) / 8);
    scores_kernel<<<sblocks, 256, 0, stream>>>(h, a, s_src, s_dst, Nn);

    int tot_e = En + Nn;
    max_kernel<<<(tot_e + 255) / 256, 256, 0, stream>>>(rowp, colp, s_src, s_dst, lmax, Nn, En);
    denom_kernel<<<(tot_e + 255) / 256, 256, 0, stream>>>(rowp, colp, s_src, s_dst, lmax, denom, Nn, En);
    scatter_kernel<<<tot_e, DFEAT, 0, stream>>>(rowp, colp, h, s_src, s_dst, lmax, denom, e, out, Nn, En);
}
